// GATRegressorDO_58755152609416
// MI455X (gfx1250) — matrix operand order repaired and checked by reading, not compiled, not run on hardware
//
#include <hip/hip_runtime.h>
#include <hip/hip_bf16.h>
#include <math.h>

#define NN      100000     // nodes
#define NG      256        // graphs
#define FIN     16
#define HID     64
#define NSLOPE  0.2f
#define ORD_NEG_INF 0x007FFFFFu   // f2ord(-inf)

typedef __attribute__((ext_vector_type(16))) _Float16 v16h;
typedef __attribute__((ext_vector_type(8)))  float    v8f;

// order-preserving float<->uint encoding for atomicMax on floats
__device__ __forceinline__ unsigned f2ord(float f) {
  unsigned u = __float_as_uint(f);
  return (u & 0x80000000u) ? ~u : (u | 0x80000000u);
}
__device__ __forceinline__ float ord2f(unsigned u) {
  unsigned v = (u & 0x80000000u) ? (u ^ 0x80000000u) : ~u;
  return __uint_as_float(v);
}

// ---------------------------------------------------------------------------
// Fold attention vectors through W:  w_as[k] = sum_j W[k,j]*a[j]
// wfold layout: [w_as1(16) | w_ad1(16) | w_as2(64) | w_ad2(64)]
// ---------------------------------------------------------------------------
__global__ void fold_kernel(const float* __restrict__ W1,
                            const float* __restrict__ as1, const float* __restrict__ ad1,
                            const float* __restrict__ W2,
                            const float* __restrict__ as2, const float* __restrict__ ad2,
                            float* __restrict__ wfold) {
  int t = threadIdx.x;
  if (t < 16) {
    float s = 0.f;
    for (int j = 0; j < 64; ++j) s += W1[t * 64 + j] * as1[j];
    wfold[t] = s;
  } else if (t < 32) {
    int k = t - 16; float s = 0.f;
    for (int j = 0; j < 64; ++j) s += W1[k * 64 + j] * ad1[j];
    wfold[16 + k] = s;
  } else if (t < 96) {
    int k = t - 32; float s = 0.f;
    for (int j = 0; j < 64; ++j) s += W2[k * 64 + j] * as2[j];
    wfold[32 + k] = s;
  } else if (t < 160) {
    int k = t - 96; float s = 0.f;
    for (int j = 0; j < 64; ++j) s += W2[k * 64 + j] * ad2[j];
    wfold[96 + k] = s;
  }
}

// ---------------------------------------------------------------------------
// WMMA node transform: Hout[N,64] = Xin[N,K] @ Wmat[K,64], K in {16,64}.
// K is a template parameter so all pad predicates fold at compile time.
// W is converted to f16 *pre-swizzled into fragment order* in LDS, so each
// lane reads its v16h B fragment as two contiguous ds_load_b128.
// A fragment halves are 8 contiguous f32 row elements -> b128-mergeable loads.
// ---------------------------------------------------------------------------
template <int K>
__global__ void __launch_bounds__(128)
gemm_node(const float* __restrict__ Xin, const float* __restrict__ Wmat,
          float* __restrict__ Hout, int nTiles) {
  constexpr int nK = (K + 31) / 32;                   // 1 (K=16) or 2 (K=64)
  __shared__ __align__(32) _Float16 Wl[nK * 4 * 32 * 16];

  // cooperative swizzle: Wl[((kc*4+nt)*32 + lane)*16 + h] = f16(W[kk,n])
  // with n = nt*16 + (lane&15), kk = kc*32 + (h<8 ? h : h+8) + 8*(lane>>4)
  for (int idx = threadIdx.x; idx < nK * 4 * 32 * 16; idx += blockDim.x) {
    int h    = idx & 15;
    int lane = (idx >> 4) & 31;
    int nt   = (idx >> 9) & 3;
    int kc   = idx >> 11;
    int n    = nt * 16 + (lane & 15);
    int kk   = kc * 32 + ((h < 8) ? h : h + 8) + 8 * (lane >> 4);
    Wl[idx] = (_Float16)((kk < K) ? Wmat[kk * 64 + n] : 0.0f);
  }
  __syncthreads();

  int wave = threadIdx.x >> 5;
  int lane = threadIdx.x & 31;
  int tile = blockIdx.x * 4 + wave;
  if (tile >= nTiles) return;            // wave-uniform: EXEC stays all-1s for WMMA
  int node0 = tile * 16;
  int M = lane & 15;
  int g = lane >> 4;

  // A fragments: kl = (h<8 ? h : h+8) + 8*g  (ISA 7.12.2 16-bit A 16x32)
  const float* xrow = Xin + (size_t)(node0 + M) * K;
  v16h afrag[nK];
#pragma unroll
  for (int kc = 0; kc < nK; ++kc) {
#pragma unroll
    for (int h = 0; h < 16; ++h) {
      int kl = ((h < 8) ? h : h + 8) + 8 * g;
      int kk = kc * 32 + kl;
      float xv;
      if (K == 16) xv = (h < 8) ? xrow[kk] : 0.0f;   // compile-time pad
      else         xv = xrow[kk];
      afrag[kc][h] = (_Float16)xv;
    }
  }

#pragma unroll
  for (int nt = 0; nt < 4; ++nt) {       // 4 N-tiles of 16 cover HID=64
    v8f c = {};
#pragma unroll
    for (int kc = 0; kc < nK; ++kc) {
      v16h b = *(const v16h*)&Wl[((kc * 4 + nt) * 32 + lane) * 16];
      c = __builtin_amdgcn_wmma_f32_16x16x32_f16(
          false, afrag[kc], false, b, (short)0, c, false, false);
    }
    int n = nt * 16 + M;
#pragma unroll
    for (int r = 0; r < 8; ++r)          // C: N=lane%16, M=r+8*(lane/16)
      Hout[(node0 + r + 8 * g) * 64 + n] = c[r];
  }
}

// ---------------------------------------------------------------------------
// Per-node alpha for layer 1 (folded): as[n]=x[n]·w_as1, ad[n]=x[n]·w_ad1
// ---------------------------------------------------------------------------
__global__ void alpha1_kernel(const float* __restrict__ x, const float* __restrict__ wfold,
                              float* __restrict__ as, float* __restrict__ ad, int N) {
  int n = blockIdx.x * blockDim.x + threadIdx.x;
  if (n >= N) return;
  float s = 0.f, t = 0.f;
#pragma unroll
  for (int k = 0; k < FIN; ++k) {
    float xv = x[n * FIN + k];
    s += xv * wfold[k];
    t += xv * wfold[16 + k];
  }
  as[n] = s; ad[n] = t;
}

// Init: m <- ord(-inf), s <- 0, agg <- 0
__global__ void init_ms_agg(unsigned* __restrict__ m, float* __restrict__ s,
                            float* __restrict__ agg, int N, int aggLen) {
  int i = blockIdx.x * blockDim.x + threadIdx.x;
  if (i < N) { m[i] = ORD_NEG_INF; s[i] = 0.f; }
  if (i < aggLen) agg[i] = 0.f;
}

__global__ void init_pool(unsigned* __restrict__ gmax, float* __restrict__ gsum,
                          float* __restrict__ cnt) {
  int i = blockIdx.x * blockDim.x + threadIdx.x;
  if (i < NG * HID) { gmax[i] = ORD_NEG_INF; gsum[i] = 0.f; }
  if (i < NG) cnt[i] = 0.f;
}

// Edge pass 1: leaky-relu logit, segment max over dst
__global__ void edge_logits(const int* __restrict__ src, const int* __restrict__ dst,
                            const float* __restrict__ as, const float* __restrict__ ad,
                            float* __restrict__ ebuf, unsigned* __restrict__ m, int E) {
  int e = blockIdx.x * blockDim.x + threadIdx.x;
  if (e >= E) return;
  int sN = src[e], dN = dst[e];
  float v = as[sN] + ad[dN];
  v = (v > 0.f) ? v : NSLOPE * v;
  ebuf[e] = v;
  atomicMax(&m[dN], f2ord(v));
}

// Edge pass 2: ex = exp(e - m[dst]) in place, segment sum
__global__ void edge_exp(const int* __restrict__ dst, float* __restrict__ ebuf,
                         const unsigned* __restrict__ m, float* __restrict__ ssum, int E) {
  int e = blockIdx.x * blockDim.x + threadIdx.x;
  if (e >= E) return;
  int dN = dst[e];
  float ex = __expf(ebuf[e] - ord2f(m[dN]));
  ebuf[e] = ex;
  atomicAdd(&ssum[dN], ex);
}

// Edge pass 3 (the heavy one): agg[dst,:] += (ex/s[dst]) * h[src,:]
// one wave per edge, each lane carries a float2 of the 64-wide feature row
__global__ void edge_scatter(const int* __restrict__ src, const int* __restrict__ dst,
                             const float* __restrict__ ebuf, const float* __restrict__ ssum,
                             const float* __restrict__ h, float* __restrict__ agg, int E) {
  int wid  = (blockIdx.x * blockDim.x + threadIdx.x) >> 5;
  int lane = threadIdx.x & 31;
  int nw   = (gridDim.x * blockDim.x) >> 5;
  for (int e = wid; e < E; e += nw) {
    int sN = src[e], dN = dst[e];
    float alpha = ebuf[e] / ssum[dN];
    const float2 hv = *(const float2*)(h + sN * 64 + lane * 2);
    atomicAdd(&agg[dN * 64 + lane * 2 + 0], alpha * hv.x);
    atomicAdd(&agg[dN * 64 + lane * 2 + 1], alpha * hv.y);
  }
}

// Layer-1 finish: hrelu = relu(agg + b1) in place; layer-2 alphas via folds
__global__ void node_finish1(float* __restrict__ agg, const float* __restrict__ b,
                             const float* __restrict__ wfold,
                             float* __restrict__ as, float* __restrict__ ad, int N) {
  int n = blockIdx.x * blockDim.x + threadIdx.x;
  if (n >= N) return;
  float* row = agg + n * 64;
  float s = 0.f, t = 0.f;
#pragma unroll
  for (int f = 0; f < 64; ++f) {
    float v = row[f] + b[f];
    v = (v > 0.f) ? v : 0.f;
    row[f] = v;
    s += v * wfold[32 + f];
    t += v * wfold[96 + f];
  }
  as[n] = s; ad[n] = t;
}

// Layer-2 finish fused with pooling atomics (thread per node-feature)
__global__ void node_finish2_pool(const float* __restrict__ agg, const float* __restrict__ b,
                                  const int* __restrict__ batch,
                                  unsigned* __restrict__ gmax, float* __restrict__ gsum,
                                  float* __restrict__ cnt, int N) {
  int idx = blockIdx.x * blockDim.x + threadIdx.x;
  if (idx >= N * 64) return;
  int n = idx >> 6, f = idx & 63;
  float v = agg[idx] + b[f];
  v = (v > 0.f) ? v : 0.f;
  int g = batch[n];
  atomicMax(&gmax[g * 64 + f], f2ord(v));
  atomicAdd(&gsum[g * 64 + f], v);
  if (f == 0) atomicAdd(&cnt[g], 1.0f);
}

// Final: out[g] = [gmax | gmean] @ Wf + bf
__global__ void final_kernel(const unsigned* __restrict__ gmax, const float* __restrict__ gsum,
                             const float* __restrict__ cnt, const float* __restrict__ Wf,
                             const float* __restrict__ bf, float* __restrict__ out) {
  int g = threadIdx.x;
  if (g >= NG) return;
  float acc = bf[0];
  float c = fmaxf(cnt[g], 1.0f);
#pragma unroll
  for (int f = 0; f < 64; ++f) {
    float mx = ord2f(gmax[g * 64 + f]);
    if (!isfinite(mx)) mx = 0.f;
    acc += mx * Wf[f];
    acc += (gsum[g * 64 + f] / c) * Wf[64 + f];
  }
  out[g] = acc;
}

// ---------------------------------------------------------------------------
extern "C" void kernel_launch(void* const* d_in, const int* in_sizes, int n_in,
                              void* d_out, int out_size, void* d_ws, size_t ws_size,
                              hipStream_t stream) {
  const float* x      = (const float*)d_in[0];
  const int*   ei     = (const int*)  d_in[1];
  const int*   batch  = (const int*)  d_in[2];
  const float* W1     = (const float*)d_in[3];
  const float* a_src1 = (const float*)d_in[4];
  const float* a_dst1 = (const float*)d_in[5];
  const float* b1     = (const float*)d_in[6];
  const float* W2     = (const float*)d_in[7];
  const float* a_src2 = (const float*)d_in[8];
  const float* a_dst2 = (const float*)d_in[9];
  const float* b2     = (const float*)d_in[10];
  const float* Wf     = (const float*)d_in[11];
  const float* bf     = (const float*)d_in[12];
  float* out = (float*)d_out;

  const int E = in_sizes[1] / 2;
  const int* src = ei;
  const int* dst = ei + E;

  // workspace carve-up (floats)
  float* ws = (float*)d_ws;
  float*    hbuf  = ws;                        // [NN*64]
  float*    agg   = hbuf + (size_t)NN * 64;    // [NN*64]
  float*    asb   = agg  + (size_t)NN * 64;    // [NN]
  float*    adb   = asb  + NN;                 // [NN]
  unsigned* mbuf  = (unsigned*)(adb + NN);     // [NN]
  float*    sbuf  = (float*)(mbuf + NN);       // [NN]
  float*    ebuf  = sbuf + NN;                 // [E]
  unsigned* gmax  = (unsigned*)(ebuf + E);     // [NG*64]
  float*    gsum  = (float*)(gmax + NG * 64);  // [NG*64]
  float*    cnt   = gsum + NG * 64;            // [NG]
  float*    wfold = cnt + NG;                  // [160]

  const int nTiles   = NN / 16;                       // 6250
  const int gemmBlks = (nTiles + 3) / 4;
  const int nodeBlks = (NN + 255) / 256;
  const int nfBlks   = (NN * 64 + 255) / 256;
  const int edgeBlks = (E + 255) / 256;
  const int scatBlks = 4096;                          // grid-stride wave loop

  // folds + layer-1 setup
  fold_kernel<<<1, 256, 0, stream>>>(W1, a_src1, a_dst1, W2, a_src2, a_dst2, wfold);
  init_ms_agg<<<nfBlks, 256, 0, stream>>>(mbuf, sbuf, agg, NN, NN * 64);
  gemm_node<FIN><<<gemmBlks, 128, 0, stream>>>(x, W1, hbuf, nTiles);
  alpha1_kernel<<<nodeBlks, 256, 0, stream>>>(x, wfold, asb, adb, NN);

  // layer-1 edge phase
  edge_logits <<<edgeBlks, 256, 0, stream>>>(src, dst, asb, adb, ebuf, mbuf, E);
  edge_exp    <<<edgeBlks, 256, 0, stream>>>(dst, ebuf, mbuf, sbuf, E);
  edge_scatter<<<scatBlks, 256, 0, stream>>>(src, dst, ebuf, sbuf, hbuf, agg, E);

  // relu + bias + layer-2 alphas, then layer-2 transform
  node_finish1<<<nodeBlks, 256, 0, stream>>>(agg, b1, wfold, asb, adb, NN);
  gemm_node<HID><<<gemmBlks, 128, 0, stream>>>(agg, W2, hbuf, nTiles);

  // layer-2 edge phase (re-init scratch after agg has been consumed by gemm)
  init_ms_agg<<<nfBlks, 256, 0, stream>>>(mbuf, sbuf, agg, NN, NN * 64);
  init_pool<<<(NG * 64 + 255) / 256, 256, 0, stream>>>(gmax, gsum, cnt);
  edge_logits <<<edgeBlks, 256, 0, stream>>>(src, dst, asb, adb, ebuf, mbuf, E);
  edge_exp    <<<edgeBlks, 256, 0, stream>>>(dst, ebuf, mbuf, sbuf, E);
  edge_scatter<<<scatBlks, 256, 0, stream>>>(src, dst, ebuf, sbuf, hbuf, agg, E);

  // relu + pooling + final linear
  node_finish2_pool<<<nfBlks, 256, 0, stream>>>(agg, b2, batch, gmax, gsum, cnt, NN);
  final_kernel<<<1, 256, 0, stream>>>(gmax, gsum, cnt, Wf, bf, out);
}
